// TokenRoutedMLP_76209899700386
// MI455X (gfx1250) — compile-verified, matrix operand order checked by reading, not run on hardware
//
#include <hip/hip_runtime.h>

typedef __bf16 bh;
typedef __attribute__((ext_vector_type(16))) __bf16 v16bf;
typedef __attribute__((ext_vector_type(8)))  __bf16 v8bf;
typedef __attribute__((ext_vector_type(8)))  float  v8f;

#define B_  4
#define S_  4096
#define H_  1024
#define E_  8
#define IE_ 512
#define V_  100000
#define T_  (B_*S_)

// ---------------------------------------------------------------------------
// Pack fp32 weights [E][K][N] into bf16 WMMA-fragment layout:
//   dst index = (((e*(N/16) + ntile)*(K/32) + kb)*32 + lane)*16 + j
//   k   = kb*32 + (j&7) + ((j>>3)<<4) + ((lane>>4)<<3)   (16-bit A/B VGPR map)
//   col = ntile*16 + (lane&15)
// ---------------------------------------------------------------------------
__global__ __launch_bounds__(256) void pack_w(const float* __restrict__ src,
                                              bh* __restrict__ dst,
                                              int K, int N) {
  int idx = blockIdx.x * 256 + threadIdx.x;
  int total = E_ * K * N;
  if (idx >= total) return;
  int j    = idx & 15;
  int lane = (idx >> 4) & 31;
  int rest = idx >> 9;
  int nkb  = K >> 5;
  int kb   = rest % nkb; rest /= nkb;
  int ntl  = N >> 4;
  int nt   = rest % ntl;
  int e    = rest / ntl;
  int k    = kb * 32 + (j & 7) + ((j >> 3) << 4) + ((lane >> 4) << 3);
  int col  = nt * 16 + (lane & 15);
  dst[idx] = (bh)src[((size_t)e * K + k) * N + col];
}

// ---------------------------------------------------------------------------
// Routing: one wave32 per token. mu_logits[e] = dot(mu[t], mrw[e]); base gets
// +10; argmax (first max wins, matching jnp.argmax).
// ---------------------------------------------------------------------------
__global__ __launch_bounds__(256) void route_k(const float* __restrict__ mu,
                                               const float* __restrict__ mrw,
                                               const int* __restrict__ tok,
                                               int* __restrict__ eid) {
  int t = blockIdx.x * 8 + (threadIdx.x >> 5);
  int lane = threadIdx.x & 31;
  if (t >= T_) return;
  const float* m = mu + (size_t)t * H_;
  float acc[E_];
#pragma unroll
  for (int e = 0; e < E_; ++e) acc[e] = 0.0f;
  for (int k = lane; k < H_; k += 32) {
    float mv = m[k];
#pragma unroll
    for (int e = 0; e < E_; ++e) acc[e] += mv * mrw[e * H_ + k];
  }
#pragma unroll
  for (int off = 16; off > 0; off >>= 1) {
#pragma unroll
    for (int e = 0; e < E_; ++e) acc[e] += __shfl_xor(acc[e], off, 32);
  }
  if (lane == 0) {
    int id = tok[t];
    if (id < 0) id = 0;
    if (id > V_ - 1) id = V_ - 1;
    int be = id & (E_ - 1);
    acc[be] += 10.0f;
    int best = 0;
    float bv = acc[0];
#pragma unroll
    for (int e = 1; e < E_; ++e)
      if (acc[e] > bv) { bv = acc[e]; best = e; }
    eid[t] = best;
  }
}

__global__ void zero_counts_k(int* counts) {
  if (threadIdx.x < E_) counts[threadIdx.x] = 0;
}

__global__ __launch_bounds__(256) void count_k(const int* __restrict__ eid,
                                               int* __restrict__ counts) {
  int t = blockIdx.x * 256 + threadIdx.x;
  if (t < T_) atomicAdd(&counts[eid[t]], 1);
}

__global__ void scan_k(const int* __restrict__ counts,
                       int* __restrict__ offs, int* __restrict__ cursor) {
  if (threadIdx.x == 0 && blockIdx.x == 0) {
    int run = 0;
    for (int e = 0; e < E_; ++e) {
      offs[e] = run;
      cursor[e] = run;
      run += counts[e];
    }
  }
}

__global__ __launch_bounds__(256) void scatter_k(const int* __restrict__ eid,
                                                 int* __restrict__ cursor,
                                                 int* __restrict__ perm) {
  int t = blockIdx.x * 256 + threadIdx.x;
  if (t < T_) {
    int pos = atomicAdd(&cursor[eid[t]], 1);
    perm[pos] = t;
  }
}

// ---------------------------------------------------------------------------
// Main grouped SwiGLU GEMM. Block = (expert e, 16-row tile). 8 waves:
// each wave owns 64 cols of I_E for gate/up, 128 cols of H for down.
// __launch_bounds__(256, 4): >=4 waves/SIMD (<=256 VGPRs/wave, verified: no
// VGPR-MSB addressing, no spills) so L2-streamed B loads hide latency.
// ---------------------------------------------------------------------------
__global__ __launch_bounds__(256, 4) void moe_mlp(
    const float* __restrict__ x, const bh* __restrict__ wg,
    const bh* __restrict__ wu, const bh* __restrict__ wd,
    const int* __restrict__ perm, const int* __restrict__ counts,
    const int* __restrict__ offs, float* __restrict__ out) {
  int e = blockIdx.y;
  int m0 = blockIdx.x << 4;
  int cnt = counts[e];
  if (m0 >= cnt) return;

  __shared__ bh Xs[16][1032];   // 16 x H (padded rows: stride 1032 bf16)
  __shared__ bh Hs[16][520];    // 16 x I_E (padded rows: stride 520 bf16)
  __shared__ int toks[16];

  int tid = threadIdx.x;
  if (tid < 16) {
    int r = m0 + tid;
    toks[tid] = (r < cnt) ? perm[offs[e] + r] : -1;
  }
  __syncthreads();

  // Stage X tile to LDS (fp32 -> bf16), float4 vectorized, coalesced.
  for (int i = tid; i < 4096; i += 256) {
    int r = i >> 8;
    int c = (i & 255) << 2;
    int t = toks[r];
    float4 v = make_float4(0.f, 0.f, 0.f, 0.f);
    if (t >= 0) v = *(const float4*)(x + (size_t)t * H_ + c);
    Xs[r][c + 0] = (bh)v.x;
    Xs[r][c + 1] = (bh)v.y;
    Xs[r][c + 2] = (bh)v.z;
    Xs[r][c + 3] = (bh)v.w;
  }

  int lane = tid & 31;
  int wave = tid >> 5;
  int laneM = lane & 15;
  int laneHi = lane >> 4;

  // Warm L2/WGP$ with this wave's down-projection fragments while gate/up run.
  {
    const bh* pdw = wd + (size_t)(e * (H_ / 16) + wave * 8) * ((IE_ / 32) * 512);
    for (int pf = lane; pf < 8 * (IE_ / 32); pf += 32)
      __builtin_prefetch(pdw + (size_t)pf * 512, 0, 1);
  }
  __syncthreads();

  // ---- gate + up projection, fused SiLU(g)*u -> Hs ----
  for (int nt = 0; nt < 4; ++nt) {
    int ntile = wave * 4 + nt;
    v8f cg = {0.f, 0.f, 0.f, 0.f, 0.f, 0.f, 0.f, 0.f};
    v8f cu = {0.f, 0.f, 0.f, 0.f, 0.f, 0.f, 0.f, 0.f};
    const bh* pg = wg + (size_t)(e * (IE_ / 16) + ntile) * ((H_ / 32) * 512) + lane * 16;
    const bh* pu = wu + (size_t)(e * (IE_ / 16) + ntile) * ((H_ / 32) * 512) + lane * 16;
#pragma unroll 4
    for (int kb = 0; kb < H_ / 32; ++kb) {
      int k0 = kb * 32 + (laneHi << 3);
      v8bf a0 = *(const v8bf*)&Xs[laneM][k0];
      v8bf a1 = *(const v8bf*)&Xs[laneM][k0 + 16];
      v16bf a = __builtin_shufflevector(a0, a1, 0, 1, 2, 3, 4, 5, 6, 7,
                                        8, 9, 10, 11, 12, 13, 14, 15);
      v16bf bg = *(const v16bf*)(pg + (size_t)kb * 512);
      v16bf bu = *(const v16bf*)(pu + (size_t)kb * 512);
      cg = __builtin_amdgcn_wmma_f32_16x16x32_bf16(false, a, false, bg,
                                                   (short)0, cg, false, false);
      cu = __builtin_amdgcn_wmma_f32_16x16x32_bf16(false, a, false, bu,
                                                   (short)0, cu, false, false);
    }
    int ncol = ntile * 16 + laneM;
#pragma unroll
    for (int r = 0; r < 8; ++r) {
      float g = cg[r];
      float u = cu[r];
      // SiLU(g)*u with v_rcp_f32 instead of the precise-divide expansion.
      float sig = __builtin_amdgcn_rcpf(1.0f + __expf(-g));
      float h = g * sig * u;
      Hs[r + (laneHi << 3)][ncol] = (bh)h;
    }
  }
  __syncthreads();

  // ---- down projection, store fp32 to out ----
  for (int nt = 0; nt < 8; ++nt) {
    int ntile = wave * 8 + nt;
    v8f cd = {0.f, 0.f, 0.f, 0.f, 0.f, 0.f, 0.f, 0.f};
    const bh* pd = wd + (size_t)(e * (H_ / 16) + ntile) * ((IE_ / 32) * 512) + lane * 16;
#pragma unroll 4
    for (int kb = 0; kb < IE_ / 32; ++kb) {
      int k0 = kb * 32 + (laneHi << 3);
      v8bf a0 = *(const v8bf*)&Hs[laneM][k0];
      v8bf a1 = *(const v8bf*)&Hs[laneM][k0 + 16];
      v16bf a = __builtin_shufflevector(a0, a1, 0, 1, 2, 3, 4, 5, 6, 7,
                                        8, 9, 10, 11, 12, 13, 14, 15);
      v16bf bb = *(const v16bf*)(pd + (size_t)kb * 512);
      cd = __builtin_amdgcn_wmma_f32_16x16x32_bf16(false, a, false, bb,
                                                   (short)0, cd, false, false);
    }
    int ncol = ntile * 16 + laneM;
#pragma unroll
    for (int r = 0; r < 8; ++r) {
      int M = r + (laneHi << 3);
      if (m0 + M < cnt) {
        int t = toks[M];
        out[(size_t)t * H_ + ncol] = cd[r];
      }
    }
  }
}

// ---------------------------------------------------------------------------
extern "C" void kernel_launch(void* const* d_in, const int* in_sizes, int n_in,
                              void* d_out, int out_size, void* d_ws, size_t ws_size,
                              hipStream_t stream) {
  const float* hs     = (const float*)d_in[0];
  const int*   tok    = (const int*)d_in[1];
  const float* mu     = (const float*)d_in[2];
  const float* w_gate = (const float*)d_in[3];
  const float* w_up   = (const float*)d_in[4];
  const float* w_down = (const float*)d_in[5];
  const float* mrw    = (const float*)d_in[6];
  float* out = (float*)d_out;

  const size_t WGE = (size_t)E_ * H_ * IE_;  // elems per weight tensor
  bh*  wgq    = (bh*)d_ws;
  bh*  wuq    = wgq + WGE;
  bh*  wdq    = wuq + WGE;
  int* eid    = (int*)(wdq + WGE);
  int* perm   = eid + T_;
  int* counts = perm + T_;
  int* offs   = counts + E_;
  int* cursor = offs + E_;

  int packBlocks = (int)((WGE + 255) / 256);
  pack_w<<<packBlocks, 256, 0, stream>>>(w_gate, wgq, H_, IE_);
  pack_w<<<packBlocks, 256, 0, stream>>>(w_up,   wuq, H_, IE_);
  pack_w<<<packBlocks, 256, 0, stream>>>(w_down, wdq, IE_, H_);

  route_k<<<T_ / 8, 256, 0, stream>>>(mu, mrw, tok, eid);
  zero_counts_k<<<1, 32, 0, stream>>>(counts);
  count_k<<<T_ / 256, 256, 0, stream>>>(eid, counts);
  scan_k<<<1, 1, 0, stream>>>(counts, offs, cursor);
  scatter_k<<<T_ / 256, 256, 0, stream>>>(eid, cursor, perm);

  dim3 grid(T_ / 16, E_);
  moe_mlp<<<grid, 256, 0, stream>>>(hs, wgq, wuq, wdq, perm, counts, offs, out);
}
